// Diversity4_52699248721936
// MI455X (gfx1250) — matrix operand order repaired and checked by reading, not compiled, not run on hardware
//
#include <hip/hip_runtime.h>

typedef __attribute__((ext_vector_type(16))) _Float16 v16h;
typedef __attribute__((ext_vector_type(8)))  _Float16 v8h;
typedef __attribute__((ext_vector_type(4)))  _Float16 v4h;
typedef __attribute__((ext_vector_type(8)))  float    v8f;

#define C_DIM        1000
#define C_VEC4       250          // float4 chunks per row (1000/4)
#define ROW_STRIDE_H 1032         // halves; 2064B stride = 16B aligned, spreads LDS banks
#define INV_T        0.05f        // 1/T, T = 20
#define INV_C        0.001f       // mean(softmax) == 1/C exactly
#define SCALE_F      0.3f

__device__ __forceinline__ float wred_add(float v) {
#pragma unroll
  for (int off = 16; off >= 1; off >>= 1) v += __shfl_xor(v, off, 32);
  return v;
}
__device__ __forceinline__ float wred_max(float v) {
#pragma unroll
  for (int off = 16; off >= 1; off >>= 1) v = fmaxf(v, __shfl_xor(v, off, 32));
  return v;
}

// One block = one tile of 4 samples. 256 threads = 8 wave32s.
// Rows r = 0..15 map to (sample s = r>>2, model m = r&3).
__global__ __launch_bounds__(256) void div4_main(
    const float* __restrict__ o0, const float* __restrict__ o1,
    const float* __restrict__ o2, const float* __restrict__ o3,
    float* __restrict__ ws)
{
  __shared__ __align__(16) _Float16 ldsH[16 * ROW_STRIDE_H];
  __shared__ float ldsPart[8];

  const int tid  = threadIdx.x;
  const int lane = tid & 31;
  const int wave = tid >> 5;

  // ---------- phase 1: softmax(x/T) -> center -> L2-normalize; 2 rows per wave ----------
#pragma unroll
  for (int rr = 0; rr < 2; ++rr) {
    const int r = wave * 2 + rr;
    const int s = r >> 2;
    const int m = r & 3;
    const float* base = (m == 0) ? o0 : (m == 1) ? o1 : (m == 2) ? o2 : o3;
    const float4* row = (const float4*)(base + (size_t)(blockIdx.x * 4 + s) * C_DIM);

    float4 x[8];
    float xmax = -3.0e38f;
#pragma unroll
    for (int k = 0; k < 8; ++k) {                    // global_load_b128, coalesced
      const int i4 = lane + 32 * k;
      if (i4 < C_VEC4) {
        x[k] = row[i4];
        xmax = fmaxf(xmax, fmaxf(fmaxf(x[k].x, x[k].y), fmaxf(x[k].z, x[k].w)));
      } else {
        x[k] = make_float4(0.f, 0.f, 0.f, 0.f);
      }
    }
    xmax = wred_max(xmax);

    float sum = 0.f;
#pragma unroll
    for (int k = 0; k < 8; ++k) {
      if (lane + 32 * k < C_VEC4) {
        x[k].x = __expf((x[k].x - xmax) * INV_T);
        x[k].y = __expf((x[k].y - xmax) * INV_T);
        x[k].z = __expf((x[k].z - xmax) * INV_T);
        x[k].w = __expf((x[k].w - xmax) * INV_T);
        sum += (x[k].x + x[k].y) + (x[k].z + x[k].w);
      }
    }
    sum = wred_add(sum);
    const float invSum = 1.0f / sum;

    float ssq = 0.f;
#pragma unroll
    for (int k = 0; k < 8; ++k) {
      if (lane + 32 * k < C_VEC4) {
        x[k].x = x[k].x * invSum - INV_C;
        x[k].y = x[k].y * invSum - INV_C;
        x[k].z = x[k].z * invSum - INV_C;
        x[k].w = x[k].w * invSum - INV_C;
        ssq += (x[k].x * x[k].x + x[k].y * x[k].y) + (x[k].z * x[k].z + x[k].w * x[k].w);
      }
    }
    ssq = wred_add(ssq);
    const float inv = 1.0f / sqrtf(ssq);

    _Float16* dstRow = &ldsH[r * ROW_STRIDE_H];
#pragma unroll
    for (int k = 0; k < 8; ++k) {
      const int i4 = lane + 32 * k;                  // i4 <= 255 always
      v4h h = {};                                    // pad region [1000,1024) -> 0
      if (i4 < C_VEC4) {
        h[0] = (_Float16)(x[k].x * inv);
        h[1] = (_Float16)(x[k].y * inv);
        h[2] = (_Float16)(x[k].z * inv);
        h[3] = (_Float16)(x[k].w * inv);
      }
      *(v4h*)(dstRow + 4 * i4) = h;                  // ds_store_b64, conflict-free
    }
  }
  __syncthreads();

  // ---------- phase 2: 16x16 Gram via v_wmma_f32_16x16x32_f16 ----------
  // G[M][N] = dot(v_row_M, v_row_N); 4x4 diagonal blocks are the per-sample grams.
  // Wave w accumulates K-chunks {w, w+8, w+16, w+24}*32; partial grams are additive.
  const int mrow = lane & 15;      // A: M = lane%16 ; B = A^T: N = lane%16 (same LDS row!)
  const int hg   = lane >> 4;      // K half-group select
  const _Float16* rbase = &ldsH[mrow * ROW_STRIDE_H];

  v8f acc = {};
#pragma unroll
  for (int cc = 0; cc < 4; ++cc) {
    const int kb = (wave + 8 * cc) * 32;
    // A (16x32 MxK): VGPR0-3 <- K = kb+8*hg+[0..7]; VGPR4-7 <- K = kb+16+8*hg+[0..7]
    v8h a_lo = *(const v8h*)(rbase + kb + 8 * hg);
    v8h a_hi = *(const v8h*)(rbase + kb + 16 + 8 * hg);
    v16h A = __builtin_shufflevector(a_lo, a_hi,
                                     0,1,2,3,4,5,6,7,8,9,10,11,12,13,14,15);
    // B (32x16 KxN) = A^T: VGPR0-3 <- K = kb+8*hg+[0..7]; VGPR4-7 <- K = kb+16+8*hg+[0..7] of row N
    v8h b_lo = *(const v8h*)(rbase + kb + 8 * hg);
    v8h b_hi = *(const v8h*)(rbase + kb + 16 + 8 * hg);
    v16h Bm = __builtin_shufflevector(b_lo, b_hi,
                                      0,1,2,3,4,5,6,7,8,9,10,11,12,13,14,15);
    acc = __builtin_amdgcn_wmma_f32_16x16x32_f16(false, A, false, Bm, (short)0, acc,
                                                 false, false);
  }

  // C/D layout: lane l, VGPR j -> element (M = j + 8*(l>=16), N = l%16).
  // Keep entries with same sample (M>>2 == N>>2) and model pair m<n (M&3 < N&3).
  float part = 0.f;
#pragma unroll
  for (int j = 0; j < 8; ++j) {
    const int Mr = j + 8 * hg;
    const int Nc = mrow;
    if (((Mr >> 2) == (Nc >> 2)) && ((Mr & 3) < (Nc & 3))) part += acc[j];
  }
  part = wred_add(part);
  if (lane == 0) ldsPart[wave] = part;
  __syncthreads();
  if (tid == 0) {
    float t = 0.f;
#pragma unroll
    for (int w = 0; w < 8; ++w) t += ldsPart[w];
    ws[blockIdx.x] = t;                              // sum of d[b] over this tile
  }
}

// Deterministic final reduction (fixed traversal order; no atomics).
__global__ __launch_bounds__(256) void div4_reduce(const float* __restrict__ ws, int n,
                                                   float* __restrict__ out, float coef)
{
  __shared__ float sm[256];
  float p = 0.f;
  for (int i = threadIdx.x; i < n; i += 256) p += ws[i];
  sm[threadIdx.x] = p;
  __syncthreads();
  for (int s = 128; s > 0; s >>= 1) {
    if ((int)threadIdx.x < s) sm[threadIdx.x] += sm[threadIdx.x + s];
    __syncthreads();
  }
  if (threadIdx.x == 0) out[0] = coef * sm[0];
}

extern "C" void kernel_launch(void* const* d_in, const int* in_sizes, int n_in,
                              void* d_out, int out_size, void* d_ws, size_t ws_size,
                              hipStream_t stream) {
  (void)n_in; (void)out_size; (void)ws_size;
  const float* o0 = (const float*)d_in[0];
  const float* o1 = (const float*)d_in[1];
  const float* o2 = (const float*)d_in[2];
  const float* o3 = (const float*)d_in[3];
  const int B = in_sizes[4];          // = 16384 (length of targets)
  const int tiles = B / 4;            // 4 samples per workgroup
  float* ws = (float*)d_ws;

  div4_main<<<tiles, 256, 0, stream>>>(o0, o1, o2, o3, ws);
  div4_reduce<<<1, 256, 0, stream>>>(ws, tiles, (float*)d_out, SCALE_F / (float)B);
}
